// BorderBranch_51342039056385
// MI455X (gfx1250) — matrix operand order repaired and checked by reading, not compiled, not run on hardware
//
#include <hip/hip_runtime.h>
#include <hip/hip_bf16.h>

// ---------------- problem constants ----------------
#define NB     2
#define CIN    256
#define CB     128
#define HH     100
#define WW     100
#define SS     (HH * WW)      // 10000
#define O1     (CB * 5)       // 640 = 128 (cur) + 512 (ltrb)
#define K3     640
#define M3     256
#define POOLK  11             // pool_size + 1 samples per border

typedef __attribute__((ext_vector_type(16))) __bf16 v16bf;
typedef __attribute__((ext_vector_type(8)))  __bf16 v8bf;
typedef __attribute__((ext_vector_type(8)))  float  v8f;

// ---------------- kernel 1: pack weights to bf16 ----------------
// W1 [640][256]: rows 0..127 = w_cur, rows 128..639 = w_ltrb
// W3 [256][640] = w_out
__global__ void pack_weights(const float* __restrict__ w_cur,
                             const float* __restrict__ w_ltrb,
                             const float* __restrict__ w_out,
                             __bf16* __restrict__ W1,
                             __bf16* __restrict__ W3) {
  int idx = blockIdx.x * blockDim.x + threadIdx.x;
  const int n1 = O1 * CIN;          // 163840
  const int n3 = M3 * K3;           // 163840
  if (idx < n1) {
    float v = (idx < CB * CIN) ? w_cur[idx] : w_ltrb[idx - CB * CIN];
    W1[idx] = (__bf16)v;
  } else if (idx < n1 + n3) {
    int j = idx - n1;
    W3[j] = (__bf16)w_out[j];
  }
}

// ---------------- kernel 2: feature [n][c][s] f32 -> featT [n][s][c] bf16 ----
__global__ void transpose_feat(const float* __restrict__ feat,
                               __bf16* __restrict__ featT) {
  long long idx = (long long)blockIdx.x * blockDim.x + threadIdx.x;
  const long long total = (long long)NB * SS * CIN;
  if (idx >= total) return;
  int c = (int)(idx % CIN);
  int s = (int)((idx / CIN) % SS);
  int n = (int)(idx / ((long long)CIN * SS));
  featT[idx] = (__bf16)feat[((long long)n * CIN + c) * SS + s];
}

// ---------------- WMMA bf16 GEMM -------------------------------------------
// Out[n][M][S] (f32) = A[M][K] x Bt[n][S][K]^T   (+bias, optional relu)
// Block: 256 threads = 8 waves; block tile = 128 (M) x 16 (S); wave tile 16x16.
// A fragment (16x32 bf16): lane r=lane&15 holds row r; hi=lane>>4 selects
//   K halves per the gfx1250 layout (v0..3: K 0..7 / 8..15, v4..7: +16).
// B fragment: lane holds column s0+r; same K halves as A from Bt[s][k].
__global__ void gemm_bf16_wmma(const __bf16* __restrict__ A,
                               const __bf16* __restrict__ Bt,
                               float* __restrict__ Out,
                               const float* __restrict__ bias,
                               int M, int K, int S, int relu) {
  const int n    = blockIdx.z;
  const __bf16* Bb = Bt + (long long)n * S * K;
  float*        Ob = Out + (long long)n * M * S;

  const int wave = threadIdx.x >> 5;
  const int lane = threadIdx.x & 31;
  const int r    = lane & 15;
  const int hi   = lane >> 4;

  const int row0 = blockIdx.x * 128 + wave * 16;
  const int s0   = blockIdx.y * 16;

  const __bf16* arow = A + (long long)(row0 + r) * K;
  const __bf16* brow = Bb + (long long)(s0 + r) * K;

  v8f acc = {};
  for (int kk = 0; kk < K; kk += 32) {
    union { v16bf v; v8bf h[2]; } au, bu;
    au.h[0] = *(const v8bf*)(arow + kk + hi * 8);
    au.h[1] = *(const v8bf*)(arow + kk + 16 + hi * 8);
    bu.h[0] = *(const v8bf*)(brow + kk + hi * 8);
    bu.h[1] = *(const v8bf*)(brow + kk + 16 + hi * 8);
    acc = __builtin_amdgcn_wmma_f32_16x16x32_bf16(
        /*neg_a=*/false, au.v, /*neg_b=*/false, bu.v,
        /*c_mod=*/(short)0, acc, /*reuse_a=*/false, /*reuse_b=*/false);
  }

  #pragma unroll
  for (int i = 0; i < 8; i++) {
    int row = row0 + i + hi * 8;
    float v = acc[i];
    if (bias) v += bias[row];
    if (relu) v = fmaxf(v, 0.0f);
    Ob[(long long)row * S + (s0 + r)] = v;
  }
}

// ---------------- kernel 4: per-(n,channel) instance-norm stats -------------
__global__ void instnorm_stats(const float* __restrict__ X,
                               float* __restrict__ stats) {
  const int ch = blockIdx.x;                     // 0 .. NB*O1-1
  const float* x = X + (long long)ch * SS;
  float s = 0.0f, s2 = 0.0f;
  for (int i = threadIdx.x; i < SS; i += 256) {
    float v = x[i];
    s += v; s2 += v * v;
  }
  __shared__ float sh0[256], sh1[256];
  sh0[threadIdx.x] = s; sh1[threadIdx.x] = s2;
  __syncthreads();
  for (int off = 128; off > 0; off >>= 1) {
    if (threadIdx.x < off) {
      sh0[threadIdx.x] += sh0[threadIdx.x + off];
      sh1[threadIdx.x] += sh1[threadIdx.x + off];
    }
    __syncthreads();
  }
  if (threadIdx.x == 0) {
    float m   = sh0[0] * (1.0f / SS);
    float var = sh1[0] * (1.0f / SS) - m * m;
    stats[ch * 2 + 0] = m;
    stats[ch * 2 + 1] = rsqrtf(var + 1e-5f);
  }
}

// ---------------- kernel 5: normalize + relu --------------------------------
// rows 0..127 (fm_short) -> catT[n][s][512+o] bf16 ; rows 128..639 (ltrb)
// normalized in place (f32) for the border-align gather pass.
__global__ void norm_apply(float* __restrict__ X,
                           const float* __restrict__ stats,
                           __bf16* __restrict__ catT) {
  long long idx = (long long)blockIdx.x * blockDim.x + threadIdx.x;
  const long long total = (long long)NB * O1 * SS;
  if (idx >= total) return;
  int s = (int)(idx % SS);
  long long ch = idx / SS;
  int o = (int)(ch % O1);
  int n = (int)(ch / O1);
  float m = stats[ch * 2 + 0], rs = stats[ch * 2 + 1];
  float v = fmaxf((X[idx] - m) * rs, 0.0f);
  if (o < CB) {
    catT[((long long)n * SS + s) * K3 + 4 * CB + o] = (__bf16)v;
  } else {
    X[idx] = v;
  }
}

// ---------------- kernel 6: border align (mmcv semantics) -------------------
__device__ __forceinline__ float bilinear_one(const float* __restrict__ L,
                                              float x, float y) {
  bool valid = (x >= -1.0f) && (x <= (float)WW) && (y >= -1.0f) && (y <= (float)HH);
  x = fmaxf(x, 0.0f); y = fmaxf(y, 0.0f);
  float x0f = floorf(x), y0f = floorf(y);
  bool cx = x0f >= (float)(WW - 1);
  bool cy = y0f >= (float)(HH - 1);
  int x0 = cx ? (WW - 1) : (int)x0f;
  int y0 = cy ? (HH - 1) : (int)y0f;
  int x1 = min(x0 + 1, WW - 1);
  int y1 = min(y0 + 1, HH - 1);
  float xe = cx ? (float)x0 : x;
  float ye = cy ? (float)y0 : y;
  float lx = xe - (float)x0, ly = ye - (float)y0;
  float hx = 1.0f - lx,      hy = 1.0f - ly;
  float v = hy * hx * L[y0 * WW + x0] + hy * lx * L[y0 * WW + x1] +
            ly * hx * L[y1 * WW + x0] + ly * lx * L[y1 * WW + x1];
  return valid ? v : 0.0f;
}

// One thread per (n, c, p): 4 borders x 11 samples, max-pool, write into catT.
// torch's permute+reshape == flat index c*4*SS + 4p + b, so the destination
// cat row is (c*4*SS+4p+b)/SS and column is (c*4*SS+4p+b)%SS.
__global__ void border_align_kernel(const float* __restrict__ G,   // [n][640][SS]
                                    const float* __restrict__ boxes,// [n][SS][4]
                                    __bf16* __restrict__ catT) {
  const int pchunks = (SS + 255) / 256;                   // 40
  int p = (blockIdx.x % pchunks) * 256 + threadIdx.x;
  int c = (blockIdx.x / pchunks) % CB;
  int n =  blockIdx.x / (pchunks * CB);
  if (p >= SS) return;

  const float* bx = boxes + ((long long)n * SS + p) * 4;
  float x1 = bx[0], y1 = bx[1], x2 = bx[2], y2 = bx[3];
  float bw = x2 - x1, bh = y2 - y1;

  #pragma unroll
  for (int b = 0; b < 4; b++) {
    const float* L = G + ((long long)n * O1 + CB + b * CB + c) * SS;
    float mx = -3.402823466e38f;
    #pragma unroll
    for (int k = 0; k < POOLK; k++) {
      float t = (float)k * 0.1f;
      float x, y;
      if      (b == 0) { x = x1 + t * bw; y = y1; }          // top
      else if (b == 1) { x = x1;          y = y1 + t * bh; } // left
      else if (b == 2) { x = x1 + t * bw; y = y2; }          // bottom
      else             { x = x2;          y = y1 + t * bh; } // right
      mx = fmaxf(mx, bilinear_one(L, x, y));
    }
    int flat = 4 * p + b;                  // < 4*SS
    int row  = 4 * c + flat / SS;
    int col  = flat % SS;
    catT[((long long)n * SS + col) * K3 + row] = (__bf16)mx;
  }
}

// ---------------- launch ----------------------------------------------------
static inline size_t alignUp256(size_t x) { return (x + 255) & ~(size_t)255; }

extern "C" void kernel_launch(void* const* d_in, const int* in_sizes, int n_in,
                              void* d_out, int out_size, void* d_ws, size_t ws_size,
                              hipStream_t stream) {
  const float* feature = (const float*)d_in[0];
  const float* boxes   = (const float*)d_in[1];
  // d_in[2] = wh (unused by reference)
  const float* w_cur   = (const float*)d_in[3];
  const float* w_ltrb  = (const float*)d_in[5];
  const float* w_out   = (const float*)d_in[7];
  const float* b_out   = (const float*)d_in[8];
  float* out = (float*)d_out;

  // workspace carve-up (~88 MB)
  char* ws = (char*)d_ws;
  __bf16* W1     = (__bf16*)ws;  ws += alignUp256((size_t)O1 * CIN * 2);
  __bf16* W3     = (__bf16*)ws;  ws += alignUp256((size_t)M3 * K3 * 2);
  __bf16* featT  = (__bf16*)ws;  ws += alignUp256((size_t)NB * SS * CIN * 2);
  float*  g1     = (float*)ws;   ws += alignUp256((size_t)NB * O1 * SS * 4);
  float*  stats  = (float*)ws;   ws += alignUp256((size_t)NB * O1 * 2 * 4);
  __bf16* catT   = (__bf16*)ws;  ws += alignUp256((size_t)NB * SS * K3 * 2);
  (void)ws_size; (void)in_sizes; (void)n_in; (void)out_size;

  // 1. pack weights
  {
    int total = O1 * CIN + M3 * K3;
    pack_weights<<<(total + 255) / 256, 256, 0, stream>>>(w_cur, w_ltrb, w_out, W1, W3);
  }
  // 2. transpose feature to [n][s][k] bf16
  {
    long long total = (long long)NB * SS * CIN;
    transpose_feat<<<(int)((total + 255) / 256), 256, 0, stream>>>(feature, featT);
  }
  // 3. fused conv1x1 (cur + ltrb): M=640, K=256 (bias cancelled by instance norm)
  gemm_bf16_wmma<<<dim3(O1 / 128, SS / 16, NB), 256, 0, stream>>>(
      W1, featT, g1, nullptr, O1, CIN, SS, 0);
  // 4. instance-norm stats
  instnorm_stats<<<NB * O1, 256, 0, stream>>>(g1, stats);
  // 5. normalize + relu (fm_short -> catT rows 512..639; ltrb in place)
  {
    long long total = (long long)NB * O1 * SS;
    norm_apply<<<(int)((total + 255) / 256), 256, 0, stream>>>(g1, stats, catT);
  }
  // 6. border align -> catT rows 0..511
  {
    int pchunks = (SS + 255) / 256;
    border_align_kernel<<<NB * CB * pchunks, 256, 0, stream>>>(g1, boxes, catT);
  }
  // 7. final conv1x1: M=256, K=640, bias + relu -> d_out
  gemm_bf16_wmma<<<dim3(M3 / 128, SS / 16, NB), 256, 0, stream>>>(
      W3, catT, out, b_out, M3, K3, SS, 1);
}
